// GRU_74071005987293
// MI455X (gfx1250) — matrix operand order repaired and checked by reading, not compiled, not run on hardware
//
#include <hip/hip_runtime.h>

typedef __attribute__((ext_vector_type(16))) _Float16 v16h;
typedef __attribute__((ext_vector_type(8)))  float    v8f;
typedef __attribute__((ext_vector_type(8)))  uint32_t v8u;

#define WMMA_F16(A,B,C) __builtin_amdgcn_wmma_f32_16x16x32_f16(false,(A),false,(B),(short)0,(C),false,false)

#if defined(__has_builtin)
#  if __has_builtin(__builtin_amdgcn_tanhf)
#    define GRU_HAS_TANH 1
#  endif
#endif

namespace {

constexpr int kB = 4096, kT = 512, kD = 4, kH = 16, kO = 2;

__device__ __forceinline__ float ftanh(float x){
#ifdef GRU_HAS_TANH
  return __builtin_amdgcn_tanhf(x);                            // v_tanh_f32
#else
  float e = __builtin_amdgcn_exp2f(x * 2.8853900817779268f);   // exp(2x)
  return 1.0f - 2.0f * __builtin_amdgcn_rcpf(e + 1.0f);
#endif
}
__device__ __forceinline__ float fsigm(float x){
#ifdef GRU_HAS_TANH
  return 0.5f * __builtin_amdgcn_tanhf(0.5f * x) + 0.5f;
#else
  return __builtin_amdgcn_rcpf(1.0f + __builtin_amdgcn_exp2f(-1.4426950408889634f * x));
#endif
}
// single v_cvt_pk_rtz_f16_f32, stays in registers
__device__ __forceinline__ uint32_t pk2(float a, float b){
  return __builtin_bit_cast(uint32_t, __builtin_amdgcn_cvt_pkrtz(a, b));
}

// K-concatenated layout: B' is 32x16 f16 with
//   K=0..15 : layer input vector (x_t or h_{l-1}(t)) -> elements 0..7; lanes<16 units 0..7, lanes>=16 8..15
//   K=16..31: own recurrent state h_l(t-1) -> elements 8..15, same unit split; batch = lane%16
// A tiles (16x32 f16):
//   A[0] = [Wih_r | Whh_r]   -> G0 = xr+hr (+ bir+bhr via C)
//   A[1] = [Wih_z | Whh_z]   -> G1 = xz+hz (+ biz+bhz via C)
//   A[2] = [Wih_n |   0  ]   -> Xn = xn (+ bin via C)
//   A[3] = [  0   | Whh_n]   -> Hn = hn (+ bhn via C)
// Hf: own h(t-1) packed f16; every lane holds all 16 units of its batch row:
//   Hf[0..3] = units 0..7, Hf[4..7] = units 8..15. Serves as (a) the K=16..31 half (elements 8..15)
//   of the next B', (b) the K=0..15 input half (Xin) for the layer above.
// h[8]: h(t-1) in f32 D layout (lane = batch%16, element v -> unit v + 8*(lane>=16)),
//   kept separately so the gate blend needs no unpack (and no variable-index extract).
struct Layer {
  v16h A[4]; v8f C[4]; v8u Hf; float h[8];
};

__device__ __forceinline__ void init_layer(Layer& L,
                                           const float* __restrict__ Wih,
                                           const float* __restrict__ Whh,
                                           const float* __restrict__ bih,
                                           const float* __restrict__ bhh,
                                           int din, int m, bool lo){
  const int kb = lo ? 0 : 8;   // A layout: lanes<16 hold K=0..7 & 16..23; lanes>=16 K=8..15 & 24..31
#pragma unroll
  for (int q=0;q<2;++q){       // combined r and z tiles
    const int g = 16*q + m;    // lane%16 = M (gate row within tile)
    v16h a = {};
#pragma unroll
    for (int e=0;e<8;++e){
      int k = kb + e;
      a[e]   = (k < din) ? (_Float16)Wih[g*din + k] : (_Float16)0.0f;
      a[8+e] = (_Float16)Whh[g*kH + kb + e];
    }
    L.A[q] = a;
    v8f c;
#pragma unroll
    for (int v=0;v<8;++v){ int gg = 16*q + v + (lo?0:8); c[v] = bih[gg] + bhh[gg]; }
    L.C[q] = c;
  }
  {                            // split n tiles
    const int g = 32 + m;
    v16h ax = {}, ah = {};
#pragma unroll
    for (int e=0;e<8;++e){
      int k = kb + e;
      ax[e]   = (k < din) ? (_Float16)Wih[g*din + k] : (_Float16)0.0f;
      ah[8+e] = (_Float16)Whh[g*kH + kb + e];
    }
    L.A[2] = ax; L.A[3] = ah;
    v8f cx, ch;
#pragma unroll
    for (int v=0;v<8;++v){ int gg = 32 + v + (lo?0:8); cx[v] = bih[gg]; ch[v] = bhh[gg]; }
    L.C[2] = cx; L.C[3] = ch;
  }
  v8u z = {};
  L.Hf = z;                    // h0 = 0
#pragma unroll
  for (int v=0;v<8;++v) L.h[v] = 0.0f;
}

// One GRU timestep for one layer. Xin = input pack (lanes<16 read Xin[0..3], lanes>=16 Xin[4..7]).
__device__ __forceinline__ void gru_step(Layer& L, const v8u Xin, bool lo){
  v8u bp;
#pragma unroll
  for (int v=0;v<8;++v) bp[v] = (v<4) ? Xin[v + (lo?0:4)] : L.Hf[v - (lo?4:0)];   // v_cndmask, stays in VGPRs
  const v16h B = __builtin_bit_cast(v16h, bp);

  v8f G0 = WMMA_F16(L.A[0], B, L.C[0]);   // xr+hr+bir+bhr
  v8f G1 = WMMA_F16(L.A[1], B, L.C[1]);   // xz+hz+biz+bhz
  v8f Xn = WMMA_F16(L.A[2], B, L.C[2]);   // xn+bin
  v8f Hn = WMMA_F16(L.A[3], B, L.C[3]);   // hn+bhn

  float hn[8];
#pragma unroll
  for (int v=0; v<8; ++v){
    float r = fsigm(G0[v]);
    float z = fsigm(G1[v]);
    float n = ftanh(Xn[v] + r*Hn[v]);
    hn[v] = n + z*(L.h[v] - n);
    L.h[v] = hn[v];
  }

  // Rebuild full pack of h_new: every lane gets all 16 units of its batch row.
  uint32_t p[4];
#pragma unroll
  for (int j=0;j<4;++j) p[j] = pk2(hn[2*j], hn[2*j+1]);
  v8u nf;
#pragma unroll
  for (int j=0;j<4;++j){
    uint32_t qv = (uint32_t)__shfl_xor((int)p[j], 16, 32);
    nf[j]   = lo ? p[j] : qv;   // units 0..7
    nf[4+j] = lo ? qv   : p[j]; // units 8..15
  }
  L.Hf = nf;
}

} // namespace

__global__ __launch_bounds__(32)
void gru3_fused_kernel(const float* __restrict__ x,
                       const float* __restrict__ Wih0, const float* __restrict__ Whh0,
                       const float* __restrict__ bih0, const float* __restrict__ bhh0,
                       const float* __restrict__ Wih1, const float* __restrict__ Whh1,
                       const float* __restrict__ bih1, const float* __restrict__ bhh1,
                       const float* __restrict__ Wih2, const float* __restrict__ Whh2,
                       const float* __restrict__ bih2, const float* __restrict__ bhh2,
                       const float* __restrict__ fcW,  const float* __restrict__ fcb,
                       float* __restrict__ out){
  const int  lane = threadIdx.x;
  const bool lo   = lane < 16;
  const int  m    = lane & 15;
  const int  b0   = blockIdx.x * 16;

  Layer L0, L1, L2;
  init_layer(L0, Wih0, Whh0, bih0, bhh0, kD, m, lo);
  init_layer(L1, Wih1, Whh1, bih1, bhh1, kH, m, lo);
  init_layer(L2, Wih2, Whh2, bih2, bhh2, kH, m, lo);

  const float4* xrow = (const float4*)(x + (size_t)(b0 + m) * kT * kD);

  for (int t=0; t<kT; ++t){
    float4 xt = xrow[t];
    if (t+1 < kT) __builtin_prefetch(&xrow[t+1], 0, 0);
    v8u xin = {};                 // lanes<16 read xin[0..3], lanes>=16 xin[4..7] (zero)
    xin[0] = pk2(xt.x, xt.y);
    xin[1] = pk2(xt.z, xt.w);

    gru_step(L0, xin,   lo);
    gru_step(L1, L0.Hf, lo);
    gru_step(L2, L1.Hf, lo);
  }

  // Final FC: out[b, o] = sum_j h2[b,j] * fcW[o,j] + fcb[o]
  float acc[kO];
#pragma unroll
  for (int o=0;o<kO;++o){
    float s = 0.0f;
#pragma unroll
    for (int v=0;v<8;++v){
      int j = v + (lo?0:8);       // lane-dependent *address* only (plain load)
      s += L2.h[v] * fcW[o*kH + j];
    }
    s += __shfl_xor(s, 16, 32);   // combine the two half-wave partials
    acc[o] = s + fcb[o];
  }
  if (lo){
#pragma unroll
    for (int o=0;o<kO;++o) out[(size_t)(b0+m)*kO + o] = acc[o];
  }
}

extern "C" void kernel_launch(void* const* d_in, const int* in_sizes, int n_in,
                              void* d_out, int out_size, void* d_ws, size_t ws_size,
                              hipStream_t stream) {
  (void)in_sizes; (void)n_in; (void)d_ws; (void)ws_size; (void)out_size;
  const float* x    = (const float*)d_in[0];
  const float* Wih0 = (const float*)d_in[1];
  const float* Whh0 = (const float*)d_in[2];
  const float* bih0 = (const float*)d_in[3];
  const float* bhh0 = (const float*)d_in[4];
  const float* Wih1 = (const float*)d_in[5];
  const float* Whh1 = (const float*)d_in[6];
  const float* bih1 = (const float*)d_in[7];
  const float* bhh1 = (const float*)d_in[8];
  const float* Wih2 = (const float*)d_in[9];
  const float* Whh2 = (const float*)d_in[10];
  const float* bih2 = (const float*)d_in[11];
  const float* bhh2 = (const float*)d_in[12];
  const float* fcW  = (const float*)d_in[13];
  const float* fcb  = (const float*)d_in[14];
  float* out = (float*)d_out;

  dim3 grid(kB / 16, 1, 1);   // 256 single-wave workgroups, 16 batch rows each
  dim3 block(32, 1, 1);
  gru3_fused_kernel<<<grid, block, 0, stream>>>(x,
      Wih0, Whh0, bih0, bhh0,
      Wih1, Whh1, bih1, bhh1,
      Wih2, Whh2, bih2, bhh2,
      fcW, fcb, out);
}
